// Attention_50354196578449
// MI455X (gfx1250) — matrix operand order repaired and checked by reading, not compiled, not run on hardware
//
#include <hip/hip_runtime.h>
#include <math.h>

typedef __attribute__((ext_vector_type(4)))  float         f4;
typedef __attribute__((ext_vector_type(4)))  unsigned int  u32x4;
typedef __attribute__((ext_vector_type(4)))  int           i32x4;
typedef __attribute__((ext_vector_type(16))) __bf16        v16bf;
typedef __attribute__((ext_vector_type(8)))  float         v8f;

#define B_   64
#define N_   512
#define D_   1024            // MD = QD = AD
#define ROWS (B_ * N_)       // 32768
#define NEG_INF_F (-1e30f)

#if defined(__has_builtin)
#if __has_builtin(__builtin_amdgcn_global_load_async_to_lds_b128)
#define HAS_ASYNC_LDS 1
#endif
#endif
#ifndef HAS_ASYNC_LDS
#define HAS_ASYNC_LDS 0
#endif

#if HAS_ASYNC_LDS
typedef __attribute__((address_space(1))) i32x4 glb_v4i;   // global v4i
typedef __attribute__((address_space(3))) i32x4 lds_v4i;   // LDS v4i

__device__ __forceinline__ void async_b128(const void* g, void* l) {
  // signature (from compiler diagnostic): arg0 = non-const AS1 v4i*, arg1 = AS3 ptr
  __builtin_amdgcn_global_load_async_to_lds_b128((glb_v4i*)g, (lds_v4i*)l, 0, 0);
}
__device__ __forceinline__ void wait_asynccnt0() {
#if __has_builtin(__builtin_amdgcn_s_wait_asynccnt)
  __builtin_amdgcn_s_wait_asynccnt(0);
#else
  asm volatile("s_wait_asynccnt 0x0" ::: "memory");
#endif
}
#endif

// round-to-nearest-even float -> bf16
__device__ __forceinline__ unsigned short f2bf(float f) {
  unsigned int u = __float_as_uint(f);
  unsigned int r = u + 0x7FFFu + ((u >> 16) & 1u);
  return (unsigned short)(r >> 16);
}

// ---------------------------------------------------------------------------
// memory_values (fp32, row-major, K contiguous) -> bf16, 8 elems / thread
// ---------------------------------------------------------------------------
__global__ __launch_bounds__(256) void cvt_mem_kernel(const float* __restrict__ src,
                                                      unsigned short* __restrict__ dst) {
  const size_t i = (size_t)blockIdx.x * blockDim.x + threadIdx.x;   // one 8-elem chunk
  f4 x0 = ((const f4*)src)[2 * i];
  f4 x1 = ((const f4*)src)[2 * i + 1];
  u32x4 o;
  o.x = (unsigned)f2bf(x0.x) | ((unsigned)f2bf(x0.y) << 16);
  o.y = (unsigned)f2bf(x0.z) | ((unsigned)f2bf(x0.w) << 16);
  o.z = (unsigned)f2bf(x1.x) | ((unsigned)f2bf(x1.y) << 16);
  o.w = (unsigned)f2bf(x1.z) | ((unsigned)f2bf(x1.w) << 16);
  ((u32x4*)dst)[i] = o;
}

// ---------------------------------------------------------------------------
// Wh (K x A fp32) -> WhT (A x K bf16), 32x32 LDS tile transpose
// ---------------------------------------------------------------------------
__global__ __launch_bounds__(256) void cvt_whT_kernel(const float* __restrict__ Wh,
                                                      unsigned short* __restrict__ WhT) {
  __shared__ float tile[32][33];
  const int tx = threadIdx.x & 31, ty = threadIdx.x >> 5;   // ty 0..7
  const int k0 = blockIdx.x * 32, a0 = blockIdx.y * 32;
  #pragma unroll
  for (int i = 0; i < 4; ++i)
    tile[ty + 8 * i][tx] = Wh[(size_t)(k0 + ty + 8 * i) * D_ + (a0 + tx)];
  __syncthreads();
  #pragma unroll
  for (int i = 0; i < 4; ++i)
    WhT[(size_t)(a0 + ty + 8 * i) * D_ + (k0 + tx)] = f2bf(tile[tx][ty + 8 * i]);
}

// ---------------------------------------------------------------------------
// tq = query @ Ws  (64 x 1024) -- tiny, fp32 VALU
// ---------------------------------------------------------------------------
__global__ __launch_bounds__(256) void tq_kernel(const float* __restrict__ query,
                                                 const float* __restrict__ Ws,
                                                 float* __restrict__ tq) {
  const int b = blockIdx.x;
  const int c = blockIdx.y * 256 + threadIdx.x;
  float acc = 0.f;
  for (int k = 0; k < D_; ++k)
    acc = fmaf(query[b * D_ + k], Ws[(size_t)k * D_ + c], acc);
  tq[b * D_ + c] = acc;
}

// ---------------------------------------------------------------------------
// logits[r] = sum_a tanh( (mem@Wh)[r,a] + tq[b,a] ) * v[a]
// bf16 WMMA 16x16x32, fp32 accum. Block: 256 thr = 8 waves, 64 rows.
// Waves 0-3: K[0,512) + epilogue; waves 4-7: K[512,1024), acc handed over via LDS.
// A fragments live in 128 VGPRs for the whole kernel.
// B tiles: async global->LDS double buffer if available, else VGPR-staged.
// ---------------------------------------------------------------------------
__global__ __launch_bounds__(256) void logits_kernel(
    const unsigned short* __restrict__ memBF,   // (32768, 1024) bf16
    const unsigned short* __restrict__ whT,     // (1024 a, 1024 k) bf16
    const float* __restrict__ tq,               // (64, 1024)
    const float* __restrict__ v,                // (1024)
    float* __restrict__ logits)                 // (32768)
{
#if HAS_ASYNC_LDS
  __shared__ __align__(16) unsigned char Blds[2 * 32 * 1024]; // double-buffered tile
#else
  __shared__ __align__(16) unsigned char Blds[32 * 1024];     // single tile
#endif
  __shared__ __align__(16) float accXchg[4][32][8];           // 4 KB cross-wave handoff

  const int tid   = threadIdx.x;
  const int wave  = tid >> 5;
  const int lane  = tid & 31;
  const int grp   = wave & 3;                  // 16-row group
  const int khalf = wave >> 2;                 // 0: K 0..511   1: K 512..1023
  const int hi    = lane >> 4;
  const int lrow  = lane & 15;
  const int rowBase = blockIdx.x * 64 + grp * 16;
  const int bidx    = (blockIdx.x * 64) / N_;  // batch, constant per block

  // ---- A fragments: 16 rows x 512 K resident in VGPRs (ISA 16-bit A layout) ----
  u32x4 a0[16], a1[16];
  {
    const unsigned short* ap =
        memBF + (size_t)(rowBase + lrow) * D_ + khalf * 512 + hi * 8;
    #pragma unroll
    for (int k = 0; k < 16; ++k) {
      a0[k] = *(const u32x4*)(ap + k * 32);        // K 8*hi .. +8
      a1[k] = *(const u32x4*)(ap + k * 32 + 16);   // K 16+8*hi .. +8
    }
  }

  // ---- B staging: 2048 16B chunks -> 8 per thread; XOR-16B bank swizzle ----
  int ldsOff[8], srcOff[8];
  #pragma unroll
  for (int i = 0; i < 8; ++i) {
    const int c   = tid + i * 256;     // chunk id
    const int col = c >> 7;            // 0..15
    const int kk  = (c & 127) * 8;     // K element start within column
    const int ks  = kk >> 5;           // kStep 0..31
    const int rem = kk & 31;
    const int hk  = (rem >> 3) & 1;
    const int hf  = rem >> 4;
    const int L   = hk * 16 + col;     // destination lane of this data
    ldsOff[i] = ks * 1024 + L * 32 + ((hf ^ ((L >> 3) & 1)) << 4);
    srcOff[i] = col * D_ + kk;
  }

  float logit[8];
  #pragma unroll
  for (int i = 0; i < 8; ++i) logit[i] = 0.f;

  const int sw = ((lane >> 3) & 1) << 4;       // 0 or 16

#if HAS_ASYNC_LDS
  // ---- async global->LDS double-buffered pipeline ----
  #pragma unroll
  for (int i = 0; i < 8; ++i)                   // tile 0 -> buffer 0
    async_b128(whT + srcOff[i], Blds + ldsOff[i]);
  wait_asynccnt0();
  __syncthreads();

  for (int ct = 0; ct < 64; ++ct) {
    const int buf = ct & 1;
    if (ct + 1 < 64) {                          // fill the other buffer while computing
      const unsigned short* wp = whT + (size_t)(ct + 1) * 16 * D_;
      #pragma unroll
      for (int i = 0; i < 8; ++i)
        async_b128(wp + srcOff[i], Blds + (buf ^ 1) * 32768 + ldsOff[i]);
    }

    const unsigned char* bbase = Blds + buf * 32768 + (khalf * 16) * 1024 + lane * 32;
    v8f acc = {};
    #pragma unroll
    for (int k = 0; k < 16; ++k) {
      union { u32x4 u[2]; v16bf v; } A, Bf;
      A.u[0] = a0[k];
      A.u[1] = a1[k];
      const unsigned char* bp = bbase + k * 1024;
      Bf.u[0] = *(const u32x4*)(bp + sw);
      Bf.u[1] = *(const u32x4*)(bp + (16 ^ sw));
      acc = __builtin_amdgcn_wmma_f32_16x16x32_bf16(false, A.v, false, Bf.v,
                                                    (short)0, acc, false, false);
    }

    if (khalf) {
      #pragma unroll
      for (int i = 0; i < 8; ++i) accXchg[grp][lane][i] = acc[i];
    }
    __syncthreads();   // acc handoff ready; everyone done reading current buffer
    if (!khalf) {
      const int col = ct * 16 + lrow;
      const float t  = tq[bidx * D_ + col];
      const float vv = v[col];
      #pragma unroll
      for (int i = 0; i < 8; ++i) {
        const float full = acc[i] + accXchg[grp][lane][i] + t;
        logit[i] += tanhf(full) * vv;      // D layout: VGPR i -> row i + 8*hi
      }
    }
    wait_asynccnt0();  // this wave's async writes into buf^1 have landed in LDS
    __syncthreads();   // ... and are visible to all waves; accXchg reusable
  }
#else
  // ---- fallback: VGPR-staged single-buffer pipeline ----
  u32x4 pre[8];
  #pragma unroll
  for (int i = 0; i < 8; ++i) pre[i] = *(const u32x4*)(whT + srcOff[i]);  // tile 0
  #pragma unroll
  for (int i = 0; i < 8; ++i) *(u32x4*)(Blds + ldsOff[i]) = pre[i];
  __syncthreads();

  const unsigned char* bbase = Blds + (khalf * 16) * 1024 + lane * 32;

  for (int ct = 0; ct < 64; ++ct) {
    if (ct + 1 < 64) {                                         // prefetch next tile
      const unsigned short* wp = whT + (size_t)(ct + 1) * 16 * D_;
      #pragma unroll
      for (int i = 0; i < 8; ++i) pre[i] = *(const u32x4*)(wp + srcOff[i]);
    }

    v8f acc = {};
    #pragma unroll
    for (int k = 0; k < 16; ++k) {
      union { u32x4 u[2]; v16bf v; } A, Bf;
      A.u[0] = a0[k];
      A.u[1] = a1[k];
      const unsigned char* bp = bbase + k * 1024;
      Bf.u[0] = *(const u32x4*)(bp + sw);
      Bf.u[1] = *(const u32x4*)(bp + (16 ^ sw));
      acc = __builtin_amdgcn_wmma_f32_16x16x32_bf16(false, A.v, false, Bf.v,
                                                    (short)0, acc, false, false);
    }

    if (khalf) {
      #pragma unroll
      for (int i = 0; i < 8; ++i) accXchg[grp][lane][i] = acc[i];
    }
    __syncthreads();   // acc handoff ready; everyone done reading Blds
    if (!khalf) {
      const int col = ct * 16 + lrow;
      const float t  = tq[bidx * D_ + col];
      const float vv = v[col];
      #pragma unroll
      for (int i = 0; i < 8; ++i) {
        const float full = acc[i] + accXchg[grp][lane][i] + t;
        logit[i] += tanhf(full) * vv;
      }
    }
    if (ct + 1 < 64) {
      #pragma unroll
      for (int i = 0; i < 8; ++i) *(u32x4*)(Blds + ldsOff[i]) = pre[i];
    }
    __syncthreads();
  }
#endif

  if (!khalf) {        // reduce over the 16 columns held across a half-wave
    #pragma unroll
    for (int i = 0; i < 8; ++i) {
      float s = logit[i];
      s += __shfl_xor(s, 1, 32);
      s += __shfl_xor(s, 2, 32);
      s += __shfl_xor(s, 4, 32);
      s += __shfl_xor(s, 8, 32);
      if (lrow == 0) logits[rowBase + i + 8 * hi] = s;
    }
  }
}

// ---------------------------------------------------------------------------
// masked softmax over N=512 per batch row (matches reference no_cells edge case)
// ---------------------------------------------------------------------------
__global__ __launch_bounds__(512) void softmax_kernel(const float* __restrict__ logits,
                                                      const int* __restrict__ mask,
                                                      float* __restrict__ weights) {
  __shared__ float redf[16];
  __shared__ int   redi[16];
  const int b = blockIdx.x, n = threadIdx.x;
  const int wv = n >> 5, ln = n & 31;
  const int   m = mask[b * N_ + n];
  const float l = logits[b * N_ + n];

  int cnt = m;
  #pragma unroll
  for (int o = 16; o; o >>= 1) cnt += __shfl_xor(cnt, o, 32);
  if (ln == 0) redi[wv] = cnt;
  __syncthreads();
  int total = 0;
  #pragma unroll
  for (int i = 0; i < 16; ++i) total += redi[i];
  const bool nocells = (total == 0);

  const float x = (m == 0) ? NEG_INF_F : l;
  float mx = x;
  #pragma unroll
  for (int o = 16; o; o >>= 1) mx = fmaxf(mx, __shfl_xor(mx, o, 32));
  if (ln == 0) redf[wv] = mx;
  __syncthreads();
  float gmx = redf[0];
  #pragma unroll
  for (int i = 1; i < 16; ++i) gmx = fmaxf(gmx, redf[i]);

  const float e = nocells ? 0.f : expf(x - gmx);
  float s = e;
  #pragma unroll
  for (int o = 16; o; o >>= 1) s += __shfl_xor(s, o, 32);
  __syncthreads();                       // protect redf reuse
  if (ln == 0) redf[wv] = s;
  __syncthreads();
  float gs = 0.f;
  #pragma unroll
  for (int i = 0; i < 16; ++i) gs += redf[i];

  weights[b * N_ + n] = nocells ? 0.f : e / gs;
}

// ---------------------------------------------------------------------------
// context[b,d] = sum_n weights[b,n] * mem[b,n,d]  (memory-bound, fp32)
// ---------------------------------------------------------------------------
__global__ __launch_bounds__(256) void context_kernel(const float* __restrict__ w,
                                                      const float* __restrict__ mem,
                                                      float* __restrict__ out) {
  __shared__ float ws[N_];
  const int b = blockIdx.x;
  const int d = blockIdx.y * 256 + threadIdx.x;
  for (int i = threadIdx.x; i < N_; i += 256) ws[i] = w[b * N_ + i];
  __syncthreads();
  float acc = 0.f;
  const float* mp = mem + (size_t)b * N_ * D_ + d;
  for (int n = 0; n < N_; ++n)
    acc = fmaf(ws[n], mp[(size_t)n * D_], acc);
  out[b * D_ + d] = acc;
}

// ---------------------------------------------------------------------------
extern "C" void kernel_launch(void* const* d_in, const int* in_sizes, int n_in,
                              void* d_out, int out_size, void* d_ws, size_t ws_size,
                              hipStream_t stream) {
  (void)in_sizes; (void)n_in; (void)out_size; (void)ws_size;
  const float* mem   = (const float*)d_in[0];   // (64, 512, 1024) f32
  const int*   mask  = (const int*)d_in[1];     // (64, 512) i32
  const float* query = (const float*)d_in[2];   // (64, 1024) f32
  const float* Wh    = (const float*)d_in[3];   // (1024, 1024) f32
  const float* Ws    = (const float*)d_in[4];   // (1024, 1024) f32
  const float* v     = (const float*)d_in[5];   // (1024, 1) f32
  float* out = (float*)d_out;                   // (64, 1024) f32

  char* p = (char*)d_ws;
  unsigned short* memBF = (unsigned short*)p; p += (size_t)ROWS * D_ * 2;  // 64 MB
  unsigned short* whT   = (unsigned short*)p; p += (size_t)D_ * D_ * 2;    // 2 MB
  float* tqb   = (float*)p; p += (size_t)B_ * D_ * sizeof(float);
  float* logit = (float*)p; p += (size_t)ROWS * sizeof(float);
  float* wgt   = (float*)p; p += (size_t)ROWS * sizeof(float);

  cvt_mem_kernel<<<(ROWS * (D_ / 8)) / 256, 256, 0, stream>>>(mem, memBF);
  cvt_whT_kernel<<<dim3(32, 32), 256, 0, stream>>>(Wh, whT);
  tq_kernel<<<dim3(B_, D_ / 256), 256, 0, stream>>>(query, Ws, tqb);
  logits_kernel<<<ROWS / 64, 256, 0, stream>>>(memBF, whT, tqb, v, logit);
  softmax_kernel<<<B_, N_, 0, stream>>>(logit, mask, wgt);
  context_kernel<<<dim3(B_, D_ / 256), 256, 0, stream>>>(wgt, mem, out);
}
